// CausalSelfAttention_13855564497573
// MI455X (gfx1250) — matrix operand order repaired and checked by reading, not compiled, not run on hardware
//
#include <hip/hip_runtime.h>
#include <hip/hip_bf16.h>

// ---------------- types ----------------
typedef __attribute__((ext_vector_type(16))) __bf16 v16bf;
typedef __attribute__((ext_vector_type(8)))  float  v8f;
typedef __attribute__((ext_vector_type(4)))  unsigned int v4u;
typedef __attribute__((ext_vector_type(8)))  int v8i;
typedef __attribute__((ext_vector_type(4)))  int v4i;

static __device__ __forceinline__ unsigned short f2bf(float f) {
    unsigned u = __float_as_uint(f);
    unsigned r = (u + 0x7FFFu + ((u >> 16) & 1u)) >> 16;   // round-to-nearest-even
    return (unsigned short)r;
}
static __device__ __forceinline__ unsigned pack2(float x, float y) {
    return (unsigned)f2bf(x) | ((unsigned)f2bf(y) << 16);
}
static __device__ __forceinline__ float bf2f(unsigned short s) {
    return __uint_as_float((unsigned)s << 16);
}

// ---------------- problem constants ----------------
#define T_SEQ   2048
#define N_EMBD  4096
#define N_HEAD  32
#define HEAD_D  128
#define QKV_N   12288   // 3 * N_EMBD

// ---------------- f32 -> bf16 bulk convert (bandwidth pass) ----------------
__global__ __launch_bounds__(256)
void cvt_f32_bf16(const float* __restrict__ src, unsigned short* __restrict__ dst, long n)
{
    long i = ((long)blockIdx.x * 256 + threadIdx.x) * 8;
    if (i >= n) return;
    float4 a = *(const float4*)(src + i);
    float4 b = *(const float4*)(src + i + 4);
    uint4 p;
    p.x = pack2(a.x, a.y); p.y = pack2(a.z, a.w);
    p.z = pack2(b.x, b.y); p.w = pack2(b.z, b.w);
    *(uint4*)(dst + i) = p;
}

// ---------------- TDM: 2D bf16 tile -> LDS with row padding ----------------
// D# per CDNA5 ISA ch.8: count=1, type=2("image"), data_size=1(2B),
// pad_enable: pad_interval=3 (16 DWORDs = one 32-elem row), pad_amount=3 (4 DWORDs)
// => LDS row stride = 32 + 8 = 40 halfs (80B), matching the compute-side layout.
static __device__ __forceinline__ void tdm_load_tile_2d(
    const unsigned short* gsrc, unsigned lds_off, int k_elems, int rows_total,
    int tile_k, int tile_rows)
{
    unsigned long long ga = (unsigned long long)(const void*)gsrc;
    v4u g0;
    g0[0] = 1u;                                              // count=1 (valid), load
    g0[1] = lds_off;                                         // lds_addr (bytes)
    g0[2] = (unsigned)(ga & 0xFFFFFFFFull);                  // global_addr lo
    g0[3] = (unsigned)((ga >> 32) & 0x01FFFFFFull) | (2u << 30);  // addr hi | type=2
    v8i g1;
    g1[0] = (int)((1u << 16) | (1u << 20) | (3u << 22) | (3u << 25)); // dsz=2B,pad
    g1[1] = (int)(((unsigned)k_elems & 0xFFFFu) << 16);      // tensor_dim0 lo16
    g1[2] = (int)((((unsigned)k_elems >> 16) & 0xFFFFu) |
                  (((unsigned)rows_total & 0xFFFFu) << 16)); // dim0 hi | dim1 lo
    g1[3] = (int)((((unsigned)rows_total >> 16) & 0xFFFFu) |
                  (((unsigned)tile_k) << 16));               // dim1 hi | tile_dim0
    g1[4] = (int)((unsigned)tile_rows);                      // tile_dim1 (tile_dim2=0)
    g1[5] = (int)k_elems;                                    // tensor_dim0_stride lo32
    g1[6] = 0;
    g1[7] = 0;
    v4i z4 = {0, 0, 0, 0};
    v8i z8 = {0, 0, 0, 0, 0, 0, 0, 0};
    __builtin_amdgcn_tensor_load_to_lds(g0, g1, z4, z4, z8, 0);
}

// ---------------- bf16-WMMA GEMM: C[M,N] = A[M,K] * B[N,K]^T + bias ----------------
#define BM 128
#define BN 128
#define BK 32
#define LSTR 40   // LDS row stride in halfs (80B)

__global__ __launch_bounds__(256)
void gemm_bf16_wmma(const unsigned short* __restrict__ A, const unsigned short* __restrict__ B,
                    const float* __restrict__ bias, void* __restrict__ Cout,
                    int M, int N, int K, int outBf16)
{
    __shared__ unsigned short Asm[2][BM * LSTR];
    __shared__ unsigned short Bsm[2][BN * LSTR];

    const int tid  = threadIdx.x;
    const int lane = tid & 31;
    const int wid  = tid >> 5;

    const int bm = blockIdx.y * BM;
    const int bn = blockIdx.x * BN;

    // wave tile: 4 waves along M (32 rows each), 2 waves along N (64 cols each)
    const int wm = (wid & 3) * 32;
    const int wn = (wid >> 2) * 64;

    v8f acc[2][4];
#pragma unroll
    for (int i = 0; i < 2; ++i)
#pragma unroll
        for (int j = 0; j < 4; ++j)
#pragma unroll
            for (int r = 0; r < 8; ++r) acc[i][j][r] = 0.0f;

    auto issueTiles = [&](int buf, int k0) {
        tdm_load_tile_2d(A + (long)bm * K + k0,
                         (unsigned)(unsigned long long)(&Asm[buf][0]),
                         K, M, BK, BM);
        tdm_load_tile_2d(B + (long)bn * K + k0,
                         (unsigned)(unsigned long long)(&Bsm[buf][0]),
                         K, N, BK, BN);
    };

    // A fragment (16x32 bf16): lane m holds row m; K chunks {0..7,16..23} / {8..15,24..31}
    auto loadAfrag = [&](int buf, int mt) -> v16bf {
        int row  = wm + mt * 16 + (lane & 15);
        int koff = (lane < 16) ? 0 : 8;
        const unsigned short* p = &Asm[buf][row * LSTR + koff];
        union { uint4 q[2]; v16bf v; } u;
        u.q[0] = *(const uint4*)(p);
        u.q[1] = *(const uint4*)(p + 16);
        return u.v;
    };
    // B fragment (32x16 bf16): lane n holds col n; K chunks {0..7,16..23} / {8..15,24..31}
    auto loadBfrag = [&](int buf, int nt) -> v16bf {
        int row  = wn + nt * 16 + (lane & 15);
        int koff = (lane < 16) ? 0 : 8;
        const unsigned short* p = &Bsm[buf][row * LSTR + koff];
        union { uint4 q[2]; v16bf v; } u;
        u.q[0] = *(const uint4*)(p);
        u.q[1] = *(const uint4*)(p + 16);
        return u.v;
    };

    if (wid == 0) {
        issueTiles(0, 0);
        __builtin_amdgcn_s_wait_tensorcnt(0);
    }
    __syncthreads();

    const int nK = K / BK;
    for (int kk = 0; kk < nK; ++kk) {
        const int buf = kk & 1;
        if (kk + 1 < nK && wid == 0) issueTiles(buf ^ 1, (kk + 1) * BK);

        v16bf af[2], bf[4];
        af[0] = loadAfrag(buf, 0);
        af[1] = loadAfrag(buf, 1);
#pragma unroll
        for (int j = 0; j < 4; ++j) bf[j] = loadBfrag(buf, j);

#pragma unroll
        for (int i = 0; i < 2; ++i)
#pragma unroll
            for (int j = 0; j < 4; ++j)
                acc[i][j] = __builtin_amdgcn_wmma_f32_16x16x32_bf16(
                    false, af[i], false, bf[j], (short)0, acc[i][j], false, false);

        if (kk + 1 < nK && wid == 0) __builtin_amdgcn_s_wait_tensorcnt(0);
        __syncthreads();
    }

    // epilogue: + bias, store f32 or bf16
#pragma unroll
    for (int i = 0; i < 2; ++i) {
        const int rbase = bm + wm + i * 16 + ((lane >> 4) << 3);
#pragma unroll
        for (int j = 0; j < 4; ++j) {
            const int col = bn + wn + j * 16 + (lane & 15);
            const float bv = bias[col];
            if (outBf16) {
                unsigned short* C16 = (unsigned short*)Cout;
#pragma unroll
                for (int r = 0; r < 8; ++r)
                    C16[(long)(rbase + r) * N + col] = f2bf(acc[i][j][r] + bv);
            } else {
                float* C32 = (float*)Cout;
#pragma unroll
                for (int r = 0; r < 8; ++r)
                    C32[(long)(rbase + r) * N + col] = acc[i][j][r] + bv;
            }
        }
    }
}

// ---------------- RoPE + layout permute ----------------
// qkv bf16 [T][H*3*D] -> Q,K bf16 [H][T][D] (Q pre-scaled by 1/sqrt(D)), Vt bf16 [H][D][T]
__global__ __launch_bounds__(256)
void rope_permute(const unsigned short* __restrict__ qkv,
                  unsigned short* __restrict__ Q, unsigned short* __restrict__ K,
                  unsigned short* __restrict__ Vt)
{
    long idx = (long)blockIdx.x * 256 + threadIdx.x;   // over H*T*D
    const int d = (int)(idx & 127);
    const int t = (int)((idx >> 7) & (T_SEQ - 1));
    const int h = (int)(idx >> 18);

    const long base = (long)t * QKV_N + (long)h * 384;
    float q = bf2f(qkv[base + d]);
    float k = bf2f(qkv[base + 128 + d]);
    float v = bf2f(qkv[base + 256 + d]);

    if (d < 32) {
        const int i = d & 15;
        const float theta = __powf(10000.0f, -(float)(2 * i) / 32.0f);
        const float ang = (float)t * theta;
        const float c = __cosf(ang), s = __sinf(ang);
        const int dp = (d < 16) ? d + 16 : d - 16;
        const float sgn = (d < 16) ? -1.0f : 1.0f;
        const float qp = bf2f(qkv[base + dp]);
        const float kp = bf2f(qkv[base + 128 + dp]);
        q = q * c + sgn * qp * s;
        k = k * c + sgn * kp * s;
    }
    q *= 0.08838834764831845f;   // 1/sqrt(128) folded into Q

    const long o = ((long)h * T_SEQ + t) * HEAD_D + d;
    Q[o] = f2bf(q);
    K[o] = f2bf(k);
    Vt[((long)h * HEAD_D + d) * T_SEQ + t] = f2bf(v);
}

// ---------------- flash-style causal attention ----------------
__global__ __launch_bounds__(256)
void attn_wmma(const unsigned short* __restrict__ Q, const unsigned short* __restrict__ K,
               const unsigned short* __restrict__ Vt, unsigned short* __restrict__ Out)
{
    __shared__ unsigned short P[8][16 * LSTR];   // per-wave P staging (bf16)

    const int lane = threadIdx.x & 31;
    const int wid  = threadIdx.x >> 5;
    const int tile = blockIdx.x * 8 + wid;       // 0..4095
    const int h  = tile >> 7;
    const int qt = tile & 127;
    const int qbase = qt * 16;

    const unsigned short* Qh = Q  + (long)h * T_SEQ * HEAD_D;
    const unsigned short* Kh = K  + (long)h * T_SEQ * HEAD_D;
    const unsigned short* Vh = Vt + (long)h * HEAD_D * T_SEQ;

    const int bcol = lane & 15;
    const int half8 = (lane >> 4) << 3;          // 0 or 8
    const int k1 = (lane < 16) ? 0 : 8;          // A-frag K chunk start
    const int bk = (lane < 16) ? 0 : 8;          // B-frag K chunk start

    // load the 16x128 Q tile as 4 A-fragments (16x32 each)
    v16bf qf[4];
    {
        const unsigned short* qrow = Qh + (long)(qbase + bcol) * HEAD_D;
#pragma unroll
        for (int s = 0; s < 4; ++s) {
            const unsigned short* p = qrow + s * 32 + k1;
            union { uint4 q[2]; v16bf v; } u;
            u.q[0] = *(const uint4*)(p);
            u.q[1] = *(const uint4*)(p + 16);
            qf[s] = u.v;
        }
    }

    v8f o[8];
#pragma unroll
    for (int j = 0; j < 8; ++j)
#pragma unroll
        for (int r = 0; r < 8; ++r) o[j][r] = 0.0f;
    float m[8], l[8];
#pragma unroll
    for (int r = 0; r < 8; ++r) { m[r] = -3.0e38f; l[r] = 0.0f; }

    unsigned short* Pw = &P[wid][0];

    for (int kt = 0; kt < qbase + 16; kt += 32) {
        if (kt + 32 < T_SEQ)
            __builtin_prefetch((const void*)(Kh + (long)(kt + 32) * HEAD_D), 0, 1);

        v8f S0, S1;
#pragma unroll
        for (int r = 0; r < 8; ++r) { S0[r] = 0.0f; S1[r] = 0.0f; }

        // S = Q * K^T   (two 16x16 key tiles, K-dim = 128 = 4 x 32)
#pragma unroll
        for (int s = 0; s < 4; ++s) {
            const unsigned short* pb0 = Kh + (long)(kt + bcol) * HEAD_D + s * 32 + bk;
            const unsigned short* pb1 = Kh + (long)(kt + 16 + bcol) * HEAD_D + s * 32 + bk;
            union { uint4 q[2]; v16bf v; } b0, b1;
            b0.q[0] = *(const uint4*)(pb0); b0.q[1] = *(const uint4*)(pb0 + 16);
            b1.q[0] = *(const uint4*)(pb1); b1.q[1] = *(const uint4*)(pb1 + 16);
            S0 = __builtin_amdgcn_wmma_f32_16x16x32_bf16(false, qf[s], false, b0.v, (short)0, S0, false, false);
            S1 = __builtin_amdgcn_wmma_f32_16x16x32_bf16(false, qf[s], false, b1.v, (short)0, S1, false, false);
        }

        // causal mask + online softmax (row-wise across 16 lanes of the half)
#pragma unroll
        for (int r = 0; r < 8; ++r) {
            const int qrow = qbase + half8 + r;
            if (kt + bcol > qrow)      S0[r] = -1.0e30f;
            if (kt + 16 + bcol > qrow) S1[r] = -1.0e30f;

            float mx = fmaxf(S0[r], S1[r]);
#pragma unroll
            for (int off = 8; off >= 1; off >>= 1) mx = fmaxf(mx, __shfl_xor(mx, off, 32));
            const float mn = fmaxf(m[r], mx);
            const float sc = __expf(m[r] - mn);
            m[r] = mn;
            l[r] *= sc;
#pragma unroll
            for (int j = 0; j < 8; ++j) o[j][r] *= sc;

            const float p0 = __expf(S0[r] - mn);
            const float p1 = __expf(S1[r] - mn);
            float rs = p0 + p1;
#pragma unroll
            for (int off = 8; off >= 1; off >>= 1) rs += __shfl_xor(rs, off, 32);
            l[r] += rs;

            const int lrow = half8 + r;
            Pw[lrow * LSTR + bcol]      = f2bf(p0);
            Pw[lrow * LSTR + 16 + bcol] = f2bf(p1);
        }
        asm volatile("s_wait_dscnt 0" ::: "memory");

        // reload P as an A-fragment (16x32)
        union { uint4 q[2]; v16bf v; } pa;
        {
            const unsigned short* pp = Pw + bcol * LSTR + k1;
            pa.q[0] = *(const uint4*)(pp);
            pa.q[1] = *(const uint4*)(pp + 16);
        }

        // O += P * V   (V transposed in memory: column d contiguous along t)
#pragma unroll
        for (int j = 0; j < 8; ++j) {
            const unsigned short* pv = Vh + (long)(j * 16 + bcol) * T_SEQ + kt + bk;
            union { uint4 q[2]; v16bf v; } bv;
            bv.q[0] = *(const uint4*)(pv); bv.q[1] = *(const uint4*)(pv + 16);
            o[j] = __builtin_amdgcn_wmma_f32_16x16x32_bf16(false, pa.v, false, bv.v, (short)0, o[j], false, false);
        }
        asm volatile("s_wait_dscnt 0" ::: "memory");   // P reads done before next-iter stores
    }

    // normalize and store bf16 to attn_out [T][H*D]
#pragma unroll
    for (int r = 0; r < 8; ++r) {
        const float rl = 1.0f / l[r];
        const int t = qbase + half8 + r;
#pragma unroll
        for (int j = 0; j < 8; ++j)
            Out[(long)t * N_EMBD + h * HEAD_D + j * 16 + bcol] = f2bf(o[j][r] * rl);
    }
}

// ---------------- launcher ----------------
extern "C" void kernel_launch(void* const* d_in, const int* in_sizes, int n_in,
                              void* d_out, int out_size, void* d_ws, size_t ws_size,
                              hipStream_t stream)
{
    const float* x      = (const float*)d_in[0];   // [T][C]
    const float* w_attn = (const float*)d_in[1];   // [QKV_N][C]
    const float* b_attn = (const float*)d_in[2];   // [QKV_N]
    const float* w_proj = (const float*)d_in[3];   // [C][C]
    const float* b_proj = (const float*)d_in[4];   // [C]
    float* out = (float*)d_out;                    // [T][C]

    char* ws = (char*)d_ws;
    unsigned short* xb   = (unsigned short*)(ws);                       // 2048*4096
    unsigned short* wab  = (unsigned short*)(ws + 16777216);            // 12288*4096
    unsigned short* wpb  = (unsigned short*)(ws + 117440512);           // 4096*4096
    unsigned short* qkvb = (unsigned short*)(ws + 150994944);           // 2048*12288
    unsigned short* Qb   = (unsigned short*)(ws + 201326592);           // 32*2048*128
    unsigned short* Kb   = (unsigned short*)(ws + 218103808);
    unsigned short* Vtb  = (unsigned short*)(ws + 234881024);
    unsigned short* aob  = (unsigned short*)(ws + 251658240);           // 2048*4096

    // 0) bulk f32 -> bf16 conversions (bandwidth-bound prologue)
    {
        long nx = (long)T_SEQ * N_EMBD;          // 8,388,608
        long na = (long)QKV_N * N_EMBD;          // 50,331,648
        long np = (long)N_EMBD * N_EMBD;         // 16,777,216
        cvt_f32_bf16<<<(unsigned)(nx / (256 * 8)), 256, 0, stream>>>(x, xb, nx);
        cvt_f32_bf16<<<(unsigned)(na / (256 * 8)), 256, 0, stream>>>(w_attn, wab, na);
        cvt_f32_bf16<<<(unsigned)(np / (256 * 8)), 256, 0, stream>>>(w_proj, wpb, np);
    }
    // 1) qkv = x @ w_attn^T + b_attn   (bf16 out)
    {
        dim3 grid(QKV_N / BN, T_SEQ / BM);
        gemm_bf16_wmma<<<grid, 256, 0, stream>>>(xb, wab, b_attn, qkvb,
                                                 T_SEQ, QKV_N, N_EMBD, 1);
    }
    // 2) RoPE + permute to Q/K [H][T][D], Vt [H][D][T]
    {
        const long total = (long)N_HEAD * T_SEQ * HEAD_D;   // 8,388,608
        rope_permute<<<(unsigned)(total / 256), 256, 0, stream>>>(qkvb, Qb, Kb, Vtb);
    }
    // 3) causal attention -> aob [T][H*D] bf16
    {
        attn_wmma<<<(N_HEAD * (T_SEQ / 16)) / 8, 256, 0, stream>>>(Qb, Kb, Vtb, aob);
    }
    // 4) out = aob @ w_proj^T + b_proj  (f32 out)
    {
        dim3 grid(N_EMBD / BN, T_SEQ / BM);
        gemm_bf16_wmma<<<grid, 256, 0, stream>>>(aob, wpb, b_proj, out,
                                                 T_SEQ, N_EMBD, N_EMBD, 0);
    }
}
